// LiquidityResidualBackbone_3332894622338
// MI455X (gfx1250) — matrix operand order repaired and checked by reading, not compiled, not run on hardware
//
#include <hip/hip_runtime.h>
#include <hip/hip_bf16.h>

// ---------------------------------------------------------------------------
// LiquidityResidualBackbone for MI455X (gfx1250, wave32).
//
// Token-side [256x256] GEMMs are algebraically folded into a streaming,
// L2-resident token pass; all dense B=256-row algebra runs through
// v_wmma_f32_16x16x32_bf16 with LDS-swizzled fragments (ds_load_b128 reads).
// All addressing uses 32-bit element offsets (buffers < 2^31 elements).
// ---------------------------------------------------------------------------

#define DMODEL 256
#define NHEAD  8
#define DHEAD  32
#define ZDIM   768

typedef __attribute__((ext_vector_type(16))) __bf16 bf16x16;
typedef __attribute__((ext_vector_type(8)))  float  f32x8;

static __device__ __forceinline__ unsigned short f2bf_u(float v) {
  unsigned u = __float_as_uint(v);
  return (unsigned short)((u + 0x7FFFu + ((u >> 16) & 1u)) >> 16);  // RNE
}
static __device__ __forceinline__ unsigned packbf(float a, float b) {
  return (unsigned)f2bf_u(a) | ((unsigned)f2bf_u(b) << 16);
}
static __device__ __forceinline__ float dot4(float4 a, float4 b) {
  return a.x * b.x + a.y * b.y + a.z * b.z + a.w * b.w;
}

struct U32x8 { uint4 a, b; };   // 32 bytes -> bit_cast to bf16x16

// ---------------------------------------------------------------------------
// Kernel 1: segment boundaries (port_batch sorted), q_p = seed@Wq_p, and the
// folded PMA query u[h][d] = sum_dh Wk_p[d, h*32+dh] * q_p[h*32+dh].
// ---------------------------------------------------------------------------
__global__ void prep_kernel(const int* __restrict__ pb, int T, int B,
                            int* __restrict__ start,
                            const float* __restrict__ seed,
                            const float* __restrict__ Wq_p,
                            const float* __restrict__ Wk_p,
                            float* __restrict__ uG) {
  const int tid = threadIdx.x;
  if (tid < B) {                      // lower_bound(port_batch, tid)
    int lo = 0, hi = T;
    while (lo < hi) { int mid = (lo + hi) >> 1; if (pb[mid] < tid) lo = mid + 1; else hi = mid; }
    start[tid] = lo;
  }
  if (tid == 0) start[B] = T;

  __shared__ float sd[DMODEL];
  __shared__ float qp[DMODEL];
  sd[tid] = seed[tid];
  __syncthreads();
  float acc = 0.f;
  for (int d = 0; d < DMODEL; ++d) acc += sd[d] * Wq_p[d * DMODEL + tid];
  qp[tid] = acc;
  __syncthreads();
  for (int h = 0; h < NHEAD; ++h) {
    float s = 0.f;
    for (int dh = 0; dh < DHEAD; ++dh)
      s += Wk_p[tid * DMODEL + h * DHEAD + dh] * qp[h * DHEAD + dh];
    uG[h * DMODEL + tid] = s;          // [H][256], h-major
  }
}

// ---------------------------------------------------------------------------
// Kernel 2: gather target rows. grid=B, block=256.
// ---------------------------------------------------------------------------
__global__ void gather_tgt_kernel(const float* __restrict__ emb,
                                  const int* __restrict__ tidx,
                                  float* __restrict__ tgt) {
  const int b = blockIdx.x, d = threadIdx.x;
  tgt[(b << 8) + d] = emb[((unsigned)tidx[b] << 8) + d];
}

// ---------------------------------------------------------------------------
// Kernel 3: bf16 WMMA GEMM  C[m,n] = A[m,:]@W[:,n] (+bias, +R residual, relu).
// block = 256 thr = 8 waves; wave w owns C tile (m0..m0+15, n0+16w..+15).
// K stepped by 64 = two 16x16x32 WMMA slices per iteration.
// LDS: A 16x64 bf16 row-major, B pre-swizzled to per-lane fragment order;
// fragment reads are 2x ds_load_b128 per slice for both operands.
// Edge tiles: clamped unconditional loads + select-to-zero; the wtrans branch
// is hoisted out of the unrolled staging loop (one uniform branch per step).
// grid = (M/16, ceil(N/128), batch) with per-batch strides.
// ---------------------------------------------------------------------------
__global__ void wmma_gemm_kernel(const float* __restrict__ A, int lda, int sA,
                                 const float* __restrict__ W, int ldw, int sW, int wtrans,
                                 const float* __restrict__ bias,
                                 const float* __restrict__ R, int ldr, int sR,
                                 float* __restrict__ C, int ldc, int sC,
                                 int M, int N, int K, int relu) {
  __shared__ unsigned Asm32[16 * 32];          // 16 rows x 64 bf16  (2 KB)
  __shared__ unsigned Bsw[2 * 8 * 32 * 8];     // [slice][wv][lane][8u] (16 KB)

  const int z = blockIdx.z;
  A += z * sA;  W += z * sW;  C += z * sC;  if (R) R += z * sR;

  const int m0 = blockIdx.x * 16, n0 = blockIdx.y * 128;
  const int tid = threadIdx.x, lane = tid & 31, wv = tid >> 5;
  const int ntile = n0 + wv * 16;
  const bool active = (ntile < N);             // wave-uniform
  const bool fastN = (n0 + 128 <= N) && (wtrans == 0);
  f32x8 acc = {};

  for (int kk = 0; kk < K; kk += 64) {
    if (fastN && (kk + 64 <= K)) {
      // ---- fast path: interior tile, unconditional coalesced loads ----
#pragma unroll
      for (int i = 0; i < 2; ++i) {            // A: 2 bf16-pairs / thread
        int p = tid * 2 + i, r = p >> 5, pc = p & 31;
        const float* ap = A + ((m0 + r) * lda + kk + pc * 2);
        Asm32[r * 32 + pc] = packbf(ap[0], ap[1]);
      }
#pragma unroll
      for (int i = 0; i < 16; ++i) {           // B: 16 bf16-pairs / thread
        int p = tid + i * 256, c = p & 127, k = (p >> 7) * 2;
        int s = k >> 5, r = k & 31;
        int ln = ((r >> 4) << 4) + (c & 15), w = c >> 4, e2 = (r & 15) >> 1;
        const float* wp = W + ((kk + k) * ldw + n0 + c);
        Bsw[((((s << 3) + w) << 5) + ln) * 8 + e2] = packbf(wp[0], wp[ldw]);
      }
    } else {
      // ---- edge path: clamped 32-bit offsets + select-to-zero ----
#pragma unroll
      for (int i = 0; i < 2; ++i) {
        int p = tid * 2 + i, r = p >> 5, pc = p & 31;
        int k0 = kk + pc * 2, k1 = k0 + 1;
        int k0c = k0 < K ? k0 : K - 1, k1c = k1 < K ? k1 : K - 1;
        int rowoff = (m0 + r) * lda;
        float v0 = A[rowoff + k0c];
        float v1 = A[rowoff + k1c];
        v0 = (k0 < K) ? v0 : 0.f;  v1 = (k1 < K) ? v1 : 0.f;
        Asm32[r * 32 + pc] = packbf(v0, v1);
      }
      if (wtrans) {                            // hoisted block-uniform branch
#pragma unroll
        for (int i = 0; i < 16; ++i) {
          int p = tid + i * 256, c = p & 127, k = (p >> 7) * 2;
          int s = k >> 5, r = k & 31;
          int ln = ((r >> 4) << 4) + (c & 15), w = c >> 4, e2 = (r & 15) >> 1;
          int n = n0 + c, nc = n < N ? n : N - 1;
          int k0 = kk + k, k1 = k0 + 1;
          int k0c = k0 < K ? k0 : K - 1, k1c = k1 < K ? k1 : K - 1;
          int rowoff = nc * ldw;
          float v0 = W[rowoff + k0c];
          float v1 = W[rowoff + k1c];
          bool nok = (n < N);
          v0 = (nok && k0 < K) ? v0 : 0.f;
          v1 = (nok && k1 < K) ? v1 : 0.f;
          Bsw[((((s << 3) + w) << 5) + ln) * 8 + e2] = packbf(v0, v1);
        }
      } else {
#pragma unroll
        for (int i = 0; i < 16; ++i) {
          int p = tid + i * 256, c = p & 127, k = (p >> 7) * 2;
          int s = k >> 5, r = k & 31;
          int ln = ((r >> 4) << 4) + (c & 15), w = c >> 4, e2 = (r & 15) >> 1;
          int n = n0 + c, nc = n < N ? n : N - 1;
          int k0 = kk + k, k1 = k0 + 1;
          int k0c = k0 < K ? k0 : K - 1, k1c = k1 < K ? k1 : K - 1;
          float v0 = W[k0c * ldw + nc];
          float v1 = W[k1c * ldw + nc];
          bool nok = (n < N);
          v0 = (nok && k0 < K) ? v0 : 0.f;
          v1 = (nok && k1 < K) ? v1 : 0.f;
          Bsw[((((s << 3) + w) << 5) + ln) * 8 + e2] = packbf(v0, v1);
        }
      }
    }
    __syncthreads();
    if (active) {
      const int am = lane & 15;
      const int kb2 = (lane >> 4) << 3;        // A kbase in uints (0 or 8)
#pragma unroll
      for (int s = 0; s < 2; ++s) {
        // A fragment: two contiguous 16B chunks per lane
        const uint4* apx = (const uint4*)&Asm32[am * 32 + s * 16 + kb2];
        U32x8 afr = { apx[0], apx[1] };        // +0 and +4 uints
        // B fragment: 32B contiguous per lane in swizzled store order
        const uint4* bpx = (const uint4*)&Bsw[((((s << 3) + wv) << 5) + lane) * 8];
        U32x8 bfr = { bpx[0], bpx[1] };
        acc = __builtin_amdgcn_wmma_f32_16x16x32_bf16(
            false, __builtin_bit_cast(bf16x16, afr),
            false, __builtin_bit_cast(bf16x16, bfr),
            (short)0, acc, false, false);
      }
    }
    __syncthreads();
  }

  const int n = ntile + (lane & 15);
  if (n < N) {
#pragma unroll
    for (int r = 0; r < 8; ++r) {
      int m = m0 + ((lane >> 4) << 3) + r;     // C/D layout: VGPR r -> M=r(+8)
      float v = acc[r];
      if (bias) v += bias[n];
      if (R)    v += R[m * ldr + n];
      if (relu) v = fmaxf(v, 0.f);
      C[m * ldc + n] = v;
    }
  }
}

// ---------------------------------------------------------------------------
// Kernel 4: streaming token pass. One workgroup per segment. Phase 1: wave-
// per-token folded logits (butterfly shfl_xor reduce), per-head max. Phase 2:
// thread-per-column exp-weighted token accumulation -> S_p, S_c.
// ---------------------------------------------------------------------------
__global__ void token_kernel(const float* __restrict__ emb,
                             const int* __restrict__ pidx,
                             const float* __restrict__ pw,
                             const float* __restrict__ uG,     // [H][256]
                             const float* __restrict__ Qt,     // [B][H][256]
                             const int* __restrict__ start,
                             float* __restrict__ lg,           // [T][16]
                             float* __restrict__ Sp,           // [B][H][256]
                             float* __restrict__ Sc) {
  const int b = blockIdx.x;
  const int s = start[b], e = start[b + 1];
  const int tid = threadIdx.x, lane = tid & 31, wv = tid >> 5;
  const float scale = 0.17677669529663687f;    // 32^-0.5

  __shared__ float mWave[8][16];
  __shared__ float mFin[16];
  __shared__ float eSh[16];
  __shared__ float sums[16];

  const float4* u4 = (const float4*)uG;
  const float4* q4 = (const float4*)(Qt + (b << 11));
  const int c0 = lane * 2;
  float4 ur[NHEAD][2], qr[NHEAD][2];
#pragma unroll
  for (int h = 0; h < NHEAD; ++h) {
    ur[h][0] = u4[h * 64 + c0]; ur[h][1] = u4[h * 64 + c0 + 1];
    qr[h][0] = q4[h * 64 + c0]; qr[h][1] = q4[h * 64 + c0 + 1];
  }

  float mx[16];
#pragma unroll
  for (int i = 0; i < 16; ++i) mx[i] = -3.0e38f;

  // -------- phase 1 --------
  for (int t = s + wv; t < e; t += 8) {
    const float4* rp = (const float4*)(emb + ((unsigned)pidx[t] << 8) + lane * 8);
    if (t + 8 < e)
      __builtin_prefetch(emb + ((unsigned)pidx[t + 8] << 8) + lane * 8, 0, 1);
    float4 x0 = rp[0], x1 = rp[1];
    float sm[16];
#pragma unroll
    for (int h = 0; h < NHEAD; ++h) {
      sm[h]     = dot4(x0, ur[h][0]) + dot4(x1, ur[h][1]);
      sm[8 + h] = dot4(x0, qr[h][0]) + dot4(x1, qr[h][1]);
    }
#pragma unroll
    for (int off = 16; off; off >>= 1)
#pragma unroll
      for (int i = 0; i < 16; ++i) sm[i] += __shfl_xor(sm[i], off, 32);
    float lw = __logf(pw[t] + 1e-8f);
#pragma unroll
    for (int i = 0; i < 16; ++i) { sm[i] = sm[i] * scale + lw; mx[i] = fmaxf(mx[i], sm[i]); }
    if (lane < 16) lg[(t << 4) + lane] = sm[lane];
  }
  __threadfence_block();
  if (lane == 0)
#pragma unroll
    for (int i = 0; i < 16; ++i) mWave[wv][i] = mx[i];
  __syncthreads();
  if (tid < 16) {
    float m = -3.0e38f;
#pragma unroll
    for (int w = 0; w < 8; ++w) m = fmaxf(m, mWave[w][tid]);
    mFin[tid] = m;
  }
  __syncthreads();

  // -------- phase 2 --------
  float accP[NHEAD] = {0, 0, 0, 0, 0, 0, 0, 0};
  float accC[NHEAD] = {0, 0, 0, 0, 0, 0, 0, 0};
  float sReg = 0.f;
  for (int t = s; t < e; ++t) {
    if (tid < 16) {
      float ev = __expf(lg[(t << 4) + tid] - mFin[tid]);
      eSh[tid] = ev;
      sReg += ev;
    }
    __syncthreads();
    float tv = emb[((unsigned)pidx[t] << 8) + tid];   // L2-resident second pass
#pragma unroll
    for (int h = 0; h < NHEAD; ++h) {
      accP[h] += eSh[h] * tv;
      accC[h] += eSh[8 + h] * tv;
    }
    __syncthreads();
  }
  if (tid < 16) sums[tid] = sReg;
  __syncthreads();
#pragma unroll
  for (int h = 0; h < NHEAD; ++h) {
    Sp[(b << 11) + (h << 8) + tid] = accP[h] / (sums[h]     + 1e-9f);
    Sc[(b << 11) + (h << 8) + tid] = accC[h] / (sums[8 + h] + 1e-9f);
  }
}

// ---------------------------------------------------------------------------
// Kernel 5: concat [tgt|contexts|fused] -> LayerNorm(768) -> zln. grid=B.
// ---------------------------------------------------------------------------
__global__ void ln_kernel(const float* __restrict__ tgt,
                          const float* __restrict__ ctx,
                          const float* __restrict__ fus,
                          const float* __restrict__ g,
                          const float* __restrict__ bta,
                          float* __restrict__ zln) {
  const int b = blockIdx.x, tid = threadIdx.x;
  __shared__ float s1[256], s2[256];
  float v0 = tgt[(b << 8) + tid];
  float v1 = ctx[(b << 8) + tid];
  float v2 = fus[(b << 8) + tid];
  s1[tid] = v0 + v1 + v2;
  s2[tid] = v0 * v0 + v1 * v1 + v2 * v2;
  __syncthreads();
  for (int o = 128; o; o >>= 1) {
    if (tid < o) { s1[tid] += s1[tid + o]; s2[tid] += s2[tid + o]; }
    __syncthreads();
  }
  float mu = s1[0] * (1.f / ZDIM);
  float var = s2[0] * (1.f / ZDIM) - mu * mu;
  float rs = rsqrtf(var + 1e-5f);
  zln[b * ZDIM + tid]       = (v0 - mu) * rs * g[tid]       + bta[tid];
  zln[b * ZDIM + 256 + tid] = (v1 - mu) * rs * g[256 + tid] + bta[256 + tid];
  zln[b * ZDIM + 512 + tid] = (v2 - mu) * rs * g[512 + tid] + bta[512 + tid];
}

// ---------------------------------------------------------------------------
static void gemm(hipStream_t st,
                 const float* A, int lda, int sA,
                 const float* W, int ldw, int sW, int wt,
                 const float* bias,
                 const float* R, int ldr, int sR,
                 float* C, int ldc, int sC,
                 int M, int N, int K, int relu, int batch) {
  dim3 grid(M / 16, (N + 127) / 128, batch);
  wmma_gemm_kernel<<<grid, 256, 0, st>>>(A, lda, sA, W, ldw, sW, wt, bias,
                                         R, ldr, sR, C, ldc, sC, M, N, K, relu);
}

extern "C" void kernel_launch(void* const* d_in, const int* in_sizes, int n_in,
                              void* d_out, int out_size, void* d_ws, size_t ws_size,
                              hipStream_t stream) {
  const float* emb   = (const float*)d_in[0];
  const int*   tidx  = (const int*)d_in[1];
  const int*   pidx  = (const int*)d_in[2];
  const int*   pb    = (const int*)d_in[3];
  const float* pw    = (const float*)d_in[4];
  const float* seed  = (const float*)d_in[5];
  const float* Wq_p  = (const float*)d_in[6];
  const float* Wk_p  = (const float*)d_in[7];
  const float* Wv_p  = (const float*)d_in[8];
  const float* Wo_p  = (const float*)d_in[9];
  const float* Wq_c  = (const float*)d_in[10];
  const float* Wk_c  = (const float*)d_in[11];
  const float* Wv_c  = (const float*)d_in[12];
  const float* Wo_c  = (const float*)d_in[13];
  const float* ln_g  = (const float*)d_in[14];
  const float* ln_b  = (const float*)d_in[15];
  const float* fW1   = (const float*)d_in[16];
  const float* fb1   = (const float*)d_in[17];
  const float* fW2   = (const float*)d_in[18];
  const float* fb2   = (const float*)d_in[19];
  const float* hW1   = (const float*)d_in[20];
  const float* hb1   = (const float*)d_in[21];
  const float* hW2   = (const float*)d_in[22];
  const float* hb2   = (const float*)d_in[23];

  const int B = in_sizes[1];
  const int T = in_sizes[2];
  const int BHD = B * NHEAD * DMODEL;

  // ---- workspace layout (floats; first 512 floats hold int start[B+1])
  int*   start = (int*)d_ws;
  float* wsf   = (float*)d_ws;
  float* uG    = wsf + 512;                  // 2048
  float* tgt   = uG + 2048;                  // B*256
  float* qc    = tgt + (size_t)B * DMODEL;   // B*256
  float* Qt    = qc + (size_t)B * DMODEL;    // B*8*256
  float* Sp    = Qt + BHD;                   // B*8*256
  float* Sc    = Sp + BHD;                   // B*8*256
  float* pooled= Sc + BHD;                   // B*256
  float* attv  = pooled + (size_t)B * DMODEL;
  float* ctx   = attv + (size_t)B * DMODEL;
  float* fus   = ctx + (size_t)B * DMODEL;
  float* zln   = fus + (size_t)B * DMODEL;   // B*768
  float* h1    = zln + (size_t)B * ZDIM;     // B*256
  float* z2    = h1 + (size_t)B * DMODEL;
  float* h2    = z2 + (size_t)B * DMODEL;
  float* lgbuf = h2 + (size_t)B * DMODEL;    // T*16

  // 1) boundaries + folded PMA query u
  prep_kernel<<<1, 256, 0, stream>>>(pb, T, B, start, seed, Wq_p, Wk_p, uG);
  // 2) gather targets
  gather_tgt_kernel<<<B, 256, 0, stream>>>(emb, tidx, tgt);
  // 3) q_c = tgt @ Wq_c
  gemm(stream, tgt, DMODEL, 0, Wq_c, DMODEL, 0, 0, nullptr, nullptr, 0, 0,
       qc, DMODEL, 0, B, DMODEL, DMODEL, 0, 1);
  // 4) Qt[b,h,:] = q_c[b,h] @ Wk_c_h^T  (batched over heads, wtrans, K=32)
  gemm(stream, qc, DMODEL, DHEAD, Wk_c, DMODEL, DHEAD, 1, nullptr, nullptr, 0, 0,
       Qt, NHEAD * DMODEL, DMODEL, B, DMODEL, DHEAD, 0, NHEAD);
  // 5) streaming token pass -> S_p, S_c
  token_kernel<<<B, 256, 0, stream>>>(emb, pidx, pw, uG, Qt, start, lgbuf, Sp, Sc);
  // 6) pooled[b,h,:] = S_p[b,h,:] @ Wv_p_h   (batched, N=32)
  gemm(stream, Sp, NHEAD * DMODEL, DMODEL, Wv_p, DMODEL, DHEAD, 0, nullptr, nullptr, 0, 0,
       pooled, DMODEL, DHEAD, B, DHEAD, DMODEL, 0, NHEAD);
  // 7) att[b,h,:] = S_c[b,h,:] @ Wv_c_h
  gemm(stream, Sc, NHEAD * DMODEL, DMODEL, Wv_c, DMODEL, DHEAD, 0, nullptr, nullptr, 0, 0,
       attv, DMODEL, DHEAD, B, DHEAD, DMODEL, 0, NHEAD);
  // 8) contexts = pooled @ Wo_p
  gemm(stream, pooled, DMODEL, 0, Wo_p, DMODEL, 0, 0, nullptr, nullptr, 0, 0,
       ctx, DMODEL, 0, B, DMODEL, DMODEL, 0, 1);
  // 9) fused = tgt + att @ Wo_c   (residual)
  gemm(stream, attv, DMODEL, 0, Wo_c, DMODEL, 0, 0, nullptr, tgt, DMODEL, 0,
       fus, DMODEL, 0, B, DMODEL, DMODEL, 0, 1);
  // 10) concat + LayerNorm
  ln_kernel<<<B, 256, 0, stream>>>(tgt, ctx, fus, ln_g, ln_b, zln);
  // 11) h1 = relu(zln @ fuse_W1 + b1)  K=768
  gemm(stream, zln, ZDIM, 0, fW1, DMODEL, 0, 0, fb1, nullptr, 0, 0,
       h1, DMODEL, 0, B, DMODEL, ZDIM, 1, 1);
  // 12) z2 = h1 @ fuse_W2 + b2
  gemm(stream, h1, DMODEL, 0, fW2, DMODEL, 0, 0, fb2, nullptr, 0, 0,
       z2, DMODEL, 0, B, DMODEL, DMODEL, 0, 1);
  // 13) h2 = relu(z2 @ head_W1 + b1)
  gemm(stream, z2, DMODEL, 0, hW1, DMODEL, 0, 0, hb1, nullptr, 0, 0,
       h2, DMODEL, 0, B, DMODEL, DMODEL, 1, 1);
  // 14) out = h2 @ head_W2 + b2   (N=3, masked 16-wide WMMA tile)
  gemm(stream, h2, DMODEL, 0, hW2, 3, 0, 0, hb2, nullptr, 0, 0,
       (float*)d_out, 3, 0, B, 3, DMODEL, 0, 1);
}
